// LSTM_14568529068425
// MI455X (gfx1250) — matrix operand order repaired and checked by reading, not compiled, not run on hardware
//
#include <hip/hip_runtime.h>
#include <hip/hip_bf16.h>
#include <math.h>

#define HD   1024
#define BT   64
#define SEQ  512
#define G4   (4 * HD)

typedef __bf16 bf16_t;
typedef __attribute__((ext_vector_type(16))) __bf16 v16bf;
typedef __attribute__((ext_vector_type(8)))  __bf16 v8bf;
typedef __attribute__((ext_vector_type(8)))  float  v8f;

#define WMMA_BF16(A, B, C) \
    __builtin_amdgcn_wmma_f32_16x16x32_bf16(false, (A), false, (B), (short)0, (C), false, false)

// Scheduler pipeline hint: 10 VMEM reads (next K-step prefetch) then 4 WMMAs.
#if __has_builtin(__builtin_amdgcn_sched_group_barrier)
#define SCHED_PIPE()                                      \
    do {                                                  \
        __builtin_amdgcn_sched_group_barrier(0x020, 10, 0); /* VMEM read */ \
        __builtin_amdgcn_sched_group_barrier(0x008, 4, 0);  /* WMMA      */ \
    } while (0)
#define SCHED_TAIL() __builtin_amdgcn_sched_group_barrier(0x008, 4, 0)
#else
#define SCHED_PIPE() do { } while (0)
#define SCHED_TAIL() do { } while (0)
#endif

__device__ __forceinline__ unsigned short f2bf(float x) {
    union { float f; unsigned u; } v; v.f = x;
    unsigned u = v.u + 0x7FFFu + ((v.u >> 16) & 1u);   // round-to-nearest-even
    return (unsigned short)(u >> 16);
}

__device__ __forceinline__ v16bf mk16(v8bf lo, v8bf hi) {
    v16bf r;
#pragma unroll
    for (int i = 0; i < 8; ++i) { r[i] = lo[i]; r[i + 8] = hi[i]; }
    return r;
}

// A-fragment, 16x32 bf16 (M x K). rowptr = base of this lane's row (m = lane&15).
// lane<16: K = k0+0..7 (v0..3), k0+16..23 (v4..7); lane>=16: +8 on both.
__device__ __forceinline__ v16bf load_a(const bf16_t* rowptr, int k0, int lane) {
    const bf16_t* p = rowptr + k0 + ((lane >> 4) << 3);
    return mk16(*(const v8bf*)p, *(const v8bf*)(p + 16));
}

// B-fragment, 32x16 bf16 (K x N). rowptr = row n (= lane&15) of W.
// lane<16: K = k0+0..7 (v0..3), k0+16..23 (v4..7); lane>=16: +8 on both.
__device__ __forceinline__ v16bf load_b(const bf16_t* rowptr, int k0, int lane) {
    const bf16_t* p = rowptr + k0 + ((lane >> 4) << 3);
    return mk16(*(const v8bf*)p, *(const v8bf*)(p + 16));
}

__device__ __forceinline__ float fsig(float x)  { return 1.f / (1.f + __expf(-x)); }
__device__ __forceinline__ float ftanh(float x) { return 1.f - 2.f / (__expf(2.f * x) + 1.f); }

__global__ void lstm_cvt_bf16(const float* __restrict__ src,
                              unsigned short* __restrict__ dst, int n) {
    int i = blockIdx.x * blockDim.x + threadIdx.x;
    int s = gridDim.x * blockDim.x;
    for (; i < n; i += s) dst[i] = f2bf(src[i]);
}

__global__ void lstm_bias_sum(const float* __restrict__ a, const float* __restrict__ b,
                              float* __restrict__ o, int n) {
    int i = blockIdx.x * blockDim.x + threadIdx.x;
    if (i < n) o[i] = a[i] + b[i];
}

__global__ void lstm_zero_u32(unsigned* __restrict__ p, int n) {
    int i = blockIdx.x * blockDim.x + threadIdx.x;
    if (i < n) p[i] = 0u;
}

// out[M x 4096] = A[M x 1024](bf16) @ W[4096 x 1024]^T (bf16) + bias, f32 accum.
// Each wave: one 16(M) x 64(N) tile = 4 accumulators sharing one A-fragment.
// Software-pipelined; sched_group_barrier pins loads-ahead-of-WMMA interleave.
__global__ __launch_bounds__(256)
void lstm_xg_gemm(const bf16_t* __restrict__ A, const bf16_t* __restrict__ W,
                  const float* __restrict__ bias, float* __restrict__ out, int M) {
    const int lane = threadIdx.x & 31;
    const int wave = blockIdx.x * (blockDim.x >> 5) + (threadIdx.x >> 5);
    const int NG = G4 / 64;                      // 64 N-groups of 64 columns
    const int mt = wave / NG;
    const int ng = wave - mt * NG;
    if (mt * 16 >= M) return;                    // wave-uniform

    const bf16_t* arow = A + (size_t)(mt * 16 + (lane & 15)) * HD;
    const bf16_t* w0   = W + (size_t)(ng * 64 + (lane & 15)) * HD;
    const bf16_t* w1   = w0 + (size_t)16 * HD;
    const bf16_t* w2   = w0 + (size_t)32 * HD;
    const bf16_t* w3   = w0 + (size_t)48 * HD;

    v8f c0 = {}, c1 = {}, c2 = {}, c3 = {};
    v16bf a  = load_a(arow, 0, lane);
    v16bf b0 = load_b(w0, 0, lane);
    v16bf b1 = load_b(w1, 0, lane);
    v16bf b2 = load_b(w2, 0, lane);
    v16bf b3 = load_b(w3, 0, lane);
#pragma unroll 2
    for (int k0 = 0; k0 < HD - 32; k0 += 32) {
        const int kn = k0 + 32;
        v16bf an  = load_a(arow, kn, lane);      // prefetch next K-step
        v16bf bn0 = load_b(w0, kn, lane);
        v16bf bn1 = load_b(w1, kn, lane);
        v16bf bn2 = load_b(w2, kn, lane);
        v16bf bn3 = load_b(w3, kn, lane);
        c0 = WMMA_BF16(a, b0, c0);
        c1 = WMMA_BF16(a, b1, c1);
        c2 = WMMA_BF16(a, b2, c2);
        c3 = WMMA_BF16(a, b3, c3);
        a = an; b0 = bn0; b1 = bn1; b2 = bn2; b3 = bn3;
        SCHED_PIPE();
    }
    c0 = WMMA_BF16(a, b0, c0);
    c1 = WMMA_BF16(a, b1, c1);
    c2 = WMMA_BF16(a, b2, c2);
    c3 = WMMA_BF16(a, b3, c3);
    SCHED_TAIL();

    const int mbase = mt * 16 + ((lane >> 4) << 3);
    const int nl    = lane & 15;
#pragma unroll
    for (int j = 0; j < 4; ++j) {
        const int n = ng * 64 + j * 16 + nl;
        const float bn = bias[n];
        v8f acc = (j == 0) ? c0 : (j == 1) ? c1 : (j == 2) ? c2 : c3;
#pragma unroll
        for (int r = 0; r < 8; ++r)
            out[(size_t)(mbase + r) * G4 + n] = acc[r] + bn;
    }
}

// Fused recurrent step: gates = h_{t-1} @ Wh^T + xg_t, then LSTM cell.
// One block = one 16x16 (m,n) tile, all 4 gates; K=1024 split over 8 waves
// (128 K each -> 4 WMMA iterations), reduced through LDS. Software-pipelined.
__global__ __launch_bounds__(256)
void lstm_step(const bf16_t* __restrict__ hbuf, unsigned short* __restrict__ hbuf_w,
               const bf16_t* __restrict__ Wh, const float* __restrict__ xg,
               float* __restrict__ cst, float* __restrict__ hout, int t) {
    __shared__ float red[8][4][256];             // 32 KB

    const int lane = threadIdx.x & 31;
    const int w    = threadIdx.x >> 5;           // wave in block: K-slice owner
    const int nt   = blockIdx.x & 63;            // 64 tiles in N (per gate block)
    const int mt   = blockIdx.x >> 6;            // 4 tiles in M (batch)
    const int m0 = mt * 16, n0 = nt * 16;
    const int kbase = w * (HD / 8);              // 128 K per wave

    const bf16_t* arow = hbuf + (size_t)t * (BT * HD) + (size_t)(m0 + (lane & 15)) * HD;
    const bf16_t* w0   = Wh + (size_t)(n0 + (lane & 15)) * HD;
    const bf16_t* w1   = w0 + (size_t)HD * HD;
    const bf16_t* w2   = w0 + (size_t)2 * HD * HD;
    const bf16_t* w3   = w0 + (size_t)3 * HD * HD;

    v8f a0 = {}, a1 = {}, a2 = {}, a3 = {};
    v16bf a  = load_a(arow, kbase, lane);
    v16bf b0 = load_b(w0, kbase, lane);
    v16bf b1 = load_b(w1, kbase, lane);
    v16bf b2 = load_b(w2, kbase, lane);
    v16bf b3 = load_b(w3, kbase, lane);
#pragma unroll
    for (int it = 0; it < 3; ++it) {
        const int kn = kbase + (it + 1) * 32;
        v16bf an  = load_a(arow, kn, lane);      // prefetch next K-step
        v16bf bn0 = load_b(w0, kn, lane);
        v16bf bn1 = load_b(w1, kn, lane);
        v16bf bn2 = load_b(w2, kn, lane);
        v16bf bn3 = load_b(w3, kn, lane);
        a0 = WMMA_BF16(a, b0, a0);
        a1 = WMMA_BF16(a, b1, a1);
        a2 = WMMA_BF16(a, b2, a2);
        a3 = WMMA_BF16(a, b3, a3);
        a = an; b0 = bn0; b1 = bn1; b2 = bn2; b3 = bn3;
        SCHED_PIPE();
    }
    a0 = WMMA_BF16(a, b0, a0);
    a1 = WMMA_BF16(a, b1, a1);
    a2 = WMMA_BF16(a, b2, a2);
    a3 = WMMA_BF16(a, b3, a3);
    SCHED_TAIL();

    // Scatter WMMA accumulators to LDS in (m,n) layout.
    const int nl = lane & 15;
    const int mh = (lane >> 4) << 3;
#pragma unroll
    for (int r = 0; r < 8; ++r) {
        const int e = (mh + r) * 16 + nl;
        red[w][0][e] = a0[r];
        red[w][1][e] = a1[r];
        red[w][2][e] = a2[r];
        red[w][3][e] = a3[r];
    }
    __syncthreads();

    // Each of the 256 threads reduces + runs the cell for one (m,n) element.
    const int tid = threadIdx.x;
    const int m = m0 + (tid >> 4);
    const int n = n0 + (tid & 15);
    float s0 = 0.f, s1 = 0.f, s2 = 0.f, s3 = 0.f;
#pragma unroll
    for (int ww = 0; ww < 8; ++ww) {
        s0 += red[ww][0][tid];
        s1 += red[ww][1][tid];
        s2 += red[ww][2][tid];
        s3 += red[ww][3][tid];
    }
    const float* xrow = xg + ((size_t)t * BT + m) * G4;
    float iv = fsig (s0 + xrow[n]);
    float fv = fsig (s1 + xrow[n + HD]);
    float gv = ftanh(s2 + xrow[n + 2 * HD]);
    float ov = fsig (s3 + xrow[n + 3 * HD]);
    float c  = fv * cst[m * HD + n] + iv * gv;
    cst[m * HD + n] = c;
    float h  = ov * ftanh(c);
    hbuf_w[(size_t)(t + 1) * (BT * HD) + (size_t)m * HD + n] = f2bf(h);
    if (hout) hout[((size_t)t * BT + m) * HD + n] = h;
}

extern "C" void kernel_launch(void* const* d_in, const int* in_sizes, int n_in,
                              void* d_out, int out_size, void* d_ws, size_t ws_size,
                              hipStream_t stream) {
    const float* X   = (const float*)d_in[0];
    const float* Wx0 = (const float*)d_in[1];
    const float* bx0 = (const float*)d_in[2];
    const float* Wh0 = (const float*)d_in[3];
    const float* bh0 = (const float*)d_in[4];
    const float* Wx1 = (const float*)d_in[5];
    const float* bx1 = (const float*)d_in[6];
    const float* Wh1 = (const float*)d_in[7];
    const float* bh1 = (const float*)d_in[8];
    float* out = (float*)d_out;

    char* ws = (char*)d_ws;
    size_t off = 0;
    auto carve = [&](size_t bytes) -> char* {
        char* p = ws + off;
        off = (off + bytes + 255) & ~(size_t)255;
        return p;
    };
    const size_t nX  = (size_t)SEQ * BT * HD;     // 33.5M elems
    const size_t nW  = (size_t)G4 * HD;           // 4.2M elems
    unsigned short* Xb   = (unsigned short*)carve(nX * 2);
    unsigned short* Wx0b = (unsigned short*)carve(nW * 2);
    unsigned short* Wh0b = (unsigned short*)carve(nW * 2);
    unsigned short* Wx1b = (unsigned short*)carve(nW * 2);
    unsigned short* Wh1b = (unsigned short*)carve(nW * 2);
    float* bias0 = (float*)carve(G4 * 4);
    float* bias1 = (float*)carve(G4 * 4);
    float* cst   = (float*)carve((size_t)BT * HD * 4);
    unsigned short* hbuf = (unsigned short*)carve((size_t)(SEQ + 1) * BT * HD * 2);
    float* xg    = (float*)carve((size_t)SEQ * BT * G4 * 4);   // 512 MB

    const int TPB = 256;
    // 1) convert inputs/weights to bf16, fold biases
    lstm_cvt_bf16<<<4096, TPB, 0, stream>>>(X, Xb, (int)nX);
    lstm_cvt_bf16<<<2048, TPB, 0, stream>>>(Wx0, Wx0b, (int)nW);
    lstm_cvt_bf16<<<2048, TPB, 0, stream>>>(Wh0, Wh0b, (int)nW);
    lstm_cvt_bf16<<<2048, TPB, 0, stream>>>(Wx1, Wx1b, (int)nW);
    lstm_cvt_bf16<<<2048, TPB, 0, stream>>>(Wh1, Wh1b, (int)nW);
    lstm_bias_sum<<<(G4 + TPB - 1) / TPB, TPB, 0, stream>>>(bx0, bh0, bias0, G4);
    lstm_bias_sum<<<(G4 + TPB - 1) / TPB, TPB, 0, stream>>>(bx1, bh1, bias1, G4);
    // zero h_{-1} slot (bf16, BT*HD elems = BT*HD/2 u32)
    lstm_zero_u32<<<(BT * HD / 2 + TPB - 1) / TPB, TPB, 0, stream>>>((unsigned*)hbuf, BT * HD / 2);

    const int M = SEQ * BT;                                   // 32768 rows
    const int gemm_blocks = (M / 16) * (G4 / 64) / 8;         // 16384 blocks of 8 waves

    // ---- layer 0 ----
    lstm_xg_gemm<<<gemm_blocks, TPB, 0, stream>>>((const bf16_t*)Xb, (const bf16_t*)Wx0b,
                                                  bias0, xg, M);
    lstm_zero_u32<<<(BT * HD + TPB - 1) / TPB, TPB, 0, stream>>>((unsigned*)cst, BT * HD);
    for (int t = 0; t < SEQ; ++t)
        lstm_step<<<256, TPB, 0, stream>>>((const bf16_t*)hbuf, hbuf, (const bf16_t*)Wh0b,
                                           xg, cst, nullptr, t);

    // ---- layer 1 ---- (input = layer-0 h, already bf16 in hbuf slots 1..SEQ)
    lstm_xg_gemm<<<gemm_blocks, TPB, 0, stream>>>((const bf16_t*)(hbuf + (size_t)BT * HD),
                                                  (const bf16_t*)Wx1b, bias1, xg, M);
    lstm_zero_u32<<<(BT * HD + TPB - 1) / TPB, TPB, 0, stream>>>((unsigned*)cst, BT * HD);
    for (int t = 0; t < SEQ; ++t)
        lstm_step<<<256, TPB, 0, stream>>>((const bf16_t*)hbuf, hbuf, (const bf16_t*)Wh1b,
                                           xg, cst, out, t);
}
